// KDNet_472446403140
// MI455X (gfx1250) — matrix operand order repaired and checked by reading, not compiled, not run on hardware
//
#include <hip/hip_runtime.h>

// ---- CDNA5 WMMA types (gfx1250, wave32) ----
typedef __bf16 bf16_t;
typedef __attribute__((ext_vector_type(16))) __bf16 v16bf;
typedef __attribute__((ext_vector_type(8)))  __bf16 v8bf;
typedef __attribute__((ext_vector_type(8)))  float  v8f;

#define BATCH 256

__device__ __forceinline__ v16bf cat8(v8bf a, v8bf b) {
    return __builtin_shufflevector(a, b, 0, 1, 2, 3, 4, 5, 6, 7,
                                         8, 9, 10, 11, 12, 13, 14, 15);
}

// Main GEMM, levels 1-9 (K % 32 == 0), all-bf16 operands, f32 accumulate:
//   Y[b, m, n] = sum_k W[m,k] * X[j,k],  j = b*D + n.
//   W: (M, K) bf16 row-major (pre-converted).  X: (B*D, K) bf16 K-contiguous.
//   Y: (B, M, D) f32 (feeds BN stats / gather).
// Wave computes 16x32: one A frag feeds two WMMAs. 4 waves/block span 128 N.
// A-frag: elem e <- K = k + 16*hi + e         => bf16 chunks {k, k+8}  from row+16*hi.
// B-frag: elem e <- K = k + 16*hi + e         => bf16 chunks {k, k+8}  from col+16*hi.
__global__ void __launch_bounds__(128)
wmma_gemm_kmul32(const bf16_t* __restrict__ W, const bf16_t* __restrict__ X,
                 float* __restrict__ Y, int M, int K, int Dbits) {
    const int lane = threadIdx.x & 31;
    const int wave = threadIdx.x >> 5;
    const int lo   = lane & 15;
    const int hi   = lane >> 4;
    const int m0   = blockIdx.x << 4;
    const int n0   = (blockIdx.y << 7) + (wave << 5);
    const int j0   = n0 + lo;
    const int j1   = j0 + 16;

    const bf16_t* wbase = W + (m0 + lo) * K + (hi << 4);
    const bf16_t* x0    = X + j0 * K + (hi << 4);
    const bf16_t* x1    = X + j1 * K + (hi << 4);

    v8f acc0 = {}, acc1 = {};
    for (int k = 0; k < K; k += 32) {
        const v16bf a  = cat8(*(const v8bf*)(wbase + k), *(const v8bf*)(wbase + k + 8));
        const v16bf b0 = cat8(*(const v8bf*)(x0 + k),    *(const v8bf*)(x0 + k + 8));
        const v16bf b1 = cat8(*(const v8bf*)(x1 + k),    *(const v8bf*)(x1 + k + 8));
        __builtin_prefetch(x0 + k + 64, 0, 1);   // global_prefetch_b8 (speculative)
        __builtin_prefetch(x1 + k + 64, 0, 1);
        acc0 = __builtin_amdgcn_wmma_f32_16x16x32_bf16(false, a, false, b0,
                                                       (short)0, acc0, false, false);
        acc1 = __builtin_amdgcn_wmma_f32_16x16x32_bf16(false, a, false, b1,
                                                       (short)0, acc1, false, false);
    }

    const int D   = 1 << Dbits;
    const int b0i = j0 >> Dbits, n0i = j0 & (D - 1);
    const int b1i = j1 >> Dbits, n1i = j1 & (D - 1);
#pragma unroll
    for (int r = 0; r < 8; ++r) {
        const int m = m0 + r + (hi << 3);          // C/D: M = r + 8*hi, N = lo
        Y[((b0i * M + m) << Dbits) + n0i] = acc0[r];
        Y[((b1i * M + m) << Dbits) + n1i] = acc1[r];
    }
}

// Level-0 GEMM (K = 3, X is the raw f32 input (B, 3, 1024)). Compile-time guards
// fold into branchless selects: only hi==0, e<3 lanes carry data; rest zeros.
__global__ void __launch_bounds__(128)
wmma_gemm_k3(const float* __restrict__ W, const float* __restrict__ X,
             float* __restrict__ Y, int M, int Dbits) {
    const int D    = 1 << Dbits;
    const int lane = threadIdx.x & 31;
    const int wave = threadIdx.x >> 5;
    const int lo   = lane & 15;
    const int hi   = lane >> 4;
    const int m0   = blockIdx.x << 4;
    const int n0   = ((blockIdx.y << 2) + wave) << 4;
    const int j    = n0 + lo;
    const int b    = j >> Dbits;
    const int n    = j & (D - 1);

    v16bf a, bf;
#pragma unroll
    for (int e = 0; e < 16; ++e) { a[e] = (bf16_t)0.0f; bf[e] = (bf16_t)0.0f; }
    const float* wrow = W + (m0 + lo) * 3;
    const float* xcol = X + ((b * 3) << Dbits) + n;
#pragma unroll
    for (int e = 0; e < 3; ++e) {
        a[e]  = (hi == 0) ? (bf16_t)wrow[e]          : (bf16_t)0.0f;
        bf[e] = (hi == 0) ? (bf16_t)xcol[e << Dbits] : (bf16_t)0.0f;
    }
    v8f acc = {};
    acc = __builtin_amdgcn_wmma_f32_16x16x32_bf16(false, a, false, bf,
                                                  (short)0, acc, false, false);
#pragma unroll
    for (int r = 0; r < 8; ++r) {
        const int m = m0 + r + (hi << 3);
        Y[((b * M + m) << Dbits) + n] = acc[r];
    }
}

// One-shot f32 -> bf16 weight staging.
__global__ void __launch_bounds__(256)
cvt_w_kernel(const float* __restrict__ w, bf16_t* __restrict__ o, int n) {
    const int i = blockIdx.x * 256 + threadIdx.x;
    if (i < n) o[i] = (bf16_t)w[i];
}

// BN stats phase 1: partial sum/sumsq per (channel, column-slice).
__global__ void __launch_bounds__(256)
bn_stats_partial(const float* __restrict__ Y, float* __restrict__ psum,
                 float* __restrict__ psq, int M, int Dbits, int chunk, int S) {
    __shared__ float s_sum[256];
    __shared__ float s_sq[256];
    const int o = blockIdx.x;
    const int s = blockIdx.y;
    const int D = 1 << Dbits;
    const int t0 = s * chunk;
    float sum = 0.f, sq = 0.f;
    for (int t = t0 + threadIdx.x; t < t0 + chunk; t += 256) {
        const int b = t >> Dbits;
        const int n = t & (D - 1);
        const float v = Y[((b * M + o) << Dbits) + n];
        sum += v; sq += v * v;
    }
    s_sum[threadIdx.x] = sum; s_sq[threadIdx.x] = sq;
    __syncthreads();
    for (int w = 128; w > 0; w >>= 1) {
        if (threadIdx.x < w) {
            s_sum[threadIdx.x] += s_sum[threadIdx.x + w];
            s_sq[threadIdx.x]  += s_sq[threadIdx.x + w];
        }
        __syncthreads();
    }
    if (threadIdx.x == 0) { psum[o * S + s] = s_sum[0]; psq[o * S + s] = s_sq[0]; }
}

// BN stats phase 2: fold slices, emit fused scale/shift (biased var = jnp.var).
__global__ void __launch_bounds__(256)
bn_stats_final(const float* __restrict__ psum, const float* __restrict__ psq,
               const float* __restrict__ g, const float* __restrict__ bb,
               float* __restrict__ scale, float* __restrict__ shift,
               int M, int S, float invBD) {
    const int o = blockIdx.x * blockDim.x + threadIdx.x;
    if (o >= M) return;
    float sum = 0.f, sq = 0.f;
    for (int s = 0; s < S; ++s) { sum += psum[o * S + s]; sq += psq[o * S + s]; }
    const float mean = sum * invBD;
    const float var  = sq * invBD - mean * mean;
    const float sc   = g[o] * rsqrtf(var + 1e-5f);
    scale[o] = sc;
    shift[o] = bb[o] - mean * sc;
}

// Fused BN + LeakyReLU + gather + repack, emitting bf16 activations in the
// (B, D/2, 2F) layout == next level's K-contiguous X. For out elem (b, p, q):
// f = q%F, n' = 2p + q/F, c = cidx[b,n'], flat = f*3D + 3n' + c, o = flat/D.
__global__ void __launch_bounds__(256)
bn_gather_kernel(const float* __restrict__ Y, const float* __restrict__ scale,
                 const float* __restrict__ shift, const int* __restrict__ cidx,
                 bf16_t* __restrict__ Xnext, int FBits, int Dbits, int total) {
    const int idx = blockIdx.x * blockDim.x + threadIdx.x;
    if (idx >= total) return;
    const int D    = 1 << Dbits;
    const int F    = 1 << FBits;
    const int q    = idx & (2 * F - 1);
    const int rest = idx >> (FBits + 1);
    const int p    = rest & ((D >> 1) - 1);
    const int b    = rest >> (Dbits - 1);
    const int f    = q & (F - 1);
    const int np   = (p << 1) + (q >> FBits);
    const int c    = cidx[(b << Dbits) + np];
    const int flat = f * 3 * D + 3 * np + c;
    const int o    = flat >> Dbits;
    const int nn   = flat & (D - 1);
    const int M    = 3 * F;
    float v = Y[((b * M + o) << Dbits) + nn];
    v = v * scale[o] + shift[o];
    v = (v >= 0.f) ? v : 0.01f * v;
    Xnext[idx] = (bf16_t)v;
}

// Final FC: out(256,40) = H(256,256) @ Wfc(40,256)^T + bfc ; H is bf16.
__global__ void __launch_bounds__(256)
fc_kernel(const bf16_t* __restrict__ H, const float* __restrict__ Wfc,
          const float* __restrict__ bfc, float* __restrict__ out) {
    const int idx = blockIdx.x * blockDim.x + threadIdx.x;
    if (idx >= BATCH * 40) return;
    const int cls = idx % 40;
    const int b   = idx / 40;
    const bf16_t* h = H + b * 256;
    const float*  w = Wfc + cls * 256;
    float acc = bfc[cls];
#pragma unroll 8
    for (int k = 0; k < 256; ++k) acc += (float)h[k] * w[k];
    out[idx] = acc;
}

extern "C" void kernel_launch(void* const* d_in, const int* in_sizes, int n_in,
                              void* d_out, int out_size, void* d_ws, size_t ws_size,
                              hipStream_t stream) {
    (void)in_sizes; (void)n_in; (void)out_size; (void)ws_size;
    static const int DIMS_[10] = {1024, 512, 256, 128, 64, 32, 16, 8, 4, 2};
    static const int FEAT_[10] = {32, 64, 64, 128, 128, 256, 256, 512, 512, 128};

    // setup_inputs() dict order: x, conv_w[0..9], bn_g[0..9], bn_b[0..9], fc_w, fc_b, c0..c9
    const float* x = (const float*)d_in[0];
    const float* conv_w[10]; const float* bn_g[10]; const float* bn_b[10];
    const int* cidx[10];
    for (int i = 0; i < 10; ++i) conv_w[i] = (const float*)d_in[1 + i];
    for (int i = 0; i < 10; ++i) bn_g[i]   = (const float*)d_in[11 + i];
    for (int i = 0; i < 10; ++i) bn_b[i]   = (const float*)d_in[21 + i];
    const float* fc_w = (const float*)d_in[31];
    const float* fc_b = (const float*)d_in[32];
    for (int i = 0; i < 10; ++i) cidx[i]   = (const int*)d_in[33 + i];

    // Workspace: Y f32 (25.2M), two bf16 act buffers (8.39M each), bf16 weights, tables
    char* ws = (char*)d_ws;
    const size_t YCAP = 25165824ull;   // 256*96*1024 f32
    const size_t ACAP = 8388608ull;    // 256*64*512 elements
    float*  Y     = (float*)ws;   ws += YCAP * sizeof(float);
    bf16_t* actA  = (bf16_t*)ws;  ws += ACAP * sizeof(bf16_t);
    bf16_t* actB  = (bf16_t*)ws;  ws += ACAP * sizeof(bf16_t);
    bf16_t* wbf   = (bf16_t*)ws;  ws += 4194304ull * sizeof(bf16_t);
    float*  scale = (float*)ws;   ws += 1536 * sizeof(float);
    float*  shift = (float*)ws;   ws += 1536 * sizeof(float);
    float*  psum  = (float*)ws;   ws += 4096 * sizeof(float);
    float*  psq   = (float*)ws;

    // Stage bf16 weights for levels 1..9 (64-element aligned offsets).
    size_t woff[10] = {0};
    {
        size_t acc = 0;
        for (int i = 1; i < 10; ++i) {
            const int nw = 3 * FEAT_[i] * 2 * FEAT_[i - 1];
            woff[i] = acc;
            acc += ((size_t)nw + 63) & ~63ull;
        }
    }
    for (int i = 1; i < 10; ++i) {
        const int nw = 3 * FEAT_[i] * 2 * FEAT_[i - 1];
        cvt_w_kernel<<<(nw + 255) / 256, 256, 0, stream>>>(conv_w[i], wbf + woff[i], nw);
    }

    const bf16_t* cur = nullptr;   // bf16 activations (levels >= 1)
    bf16_t* nxt = actA;
    for (int i = 0; i < 10; ++i) {
        const int F = FEAT_[i], Dv = DIMS_[i];
        const int K = (i == 0) ? 3 : 2 * FEAT_[i - 1];
        const int M = 3 * F;
        const int Dbits = __builtin_ctz(Dv);
        const int FBits = __builtin_ctz(F);
        const int BD = BATCH * Dv;

        if (i == 0) {
            dim3 gg(M / 16, BD / 64);
            wmma_gemm_k3<<<gg, 128, 0, stream>>>(conv_w[0], x, Y, M, Dbits);
        } else {
            dim3 gg(M / 16, BD / 128);           // BD % 128 == 0 always (BD >= 512)
            wmma_gemm_kmul32<<<gg, 128, 0, stream>>>(wbf + woff[i], cur, Y, M, K, Dbits);
        }

        const int S = (BD >= 8192) ? (BD / 8192) : 1;   // power of two, <= 32
        bn_stats_partial<<<dim3(M, S), 256, 0, stream>>>(Y, psum, psq, M, Dbits,
                                                         BD / S, S);
        bn_stats_final<<<(M + 255) / 256, 256, 0, stream>>>(psum, psq, bn_g[i],
                                                            bn_b[i], scale, shift,
                                                            M, S, 1.0f / (float)BD);

        const int total = BATCH * F * Dv;        // B * (D/2) * 2F
        bn_gather_kernel<<<(total + 255) / 256, 256, 0, stream>>>(
            Y, scale, shift, cidx[i], nxt, FBits, Dbits, total);

        cur = nxt;
        nxt = (nxt == actA) ? actB : actA;
    }

    fc_kernel<<<(BATCH * 40 + 255) / 256, 256, 0, stream>>>(cur, fc_w, fc_b,
                                                            (float*)d_out);
}
